// BirdsongLFADSModel2_34823594836032
// MI455X (gfx1250) — compile-verified, matrix operand order checked by reading, not run on hardware
//
#include <hip/hip_runtime.h>

// ---------------------------------------------------------------------------
// BirdsongLFADS on gfx1250: bf16 WMMA GEMMs (hw v_cvt_pk_bf16_f32 conversion,
// pre-converted bf16 weights, grouped/pipelined B-fragment loads, NT stores
// for the streaming logits) + persistent WMMA GRU recurrences.
// ---------------------------------------------------------------------------

typedef __attribute__((ext_vector_type(16))) __bf16   v16bf;
typedef __attribute__((ext_vector_type(8)))  unsigned v8u;
typedef __attribute__((ext_vector_type(4)))  unsigned v4u;
typedef __attribute__((ext_vector_type(8)))  float    v8f;

#define BN 32      // batch
#define TT 512     // time steps
#define HH 64      // hidden (HE=HC=HG=64)
#define G3 192     // 3*H
#define BT (BN*TT) // 16384 rows
#define HS 68      // LDS stride for h
#define GS 200     // LDS stride for gh

// Hardware packed f32->bf16 (RNE), GFX12-family VOP3: v_cvt_pk_bf16_f32
__device__ __forceinline__ unsigned pkbf(float lo, float hi) {
    unsigned r;
    asm("v_cvt_pk_bf16_f32 %0, %1, %2" : "=v"(r) : "v"(lo), "v"(hi));
    return r;
}

// 16x32 bf16 WMMA A/B fragment from row-major *f32* (ld = row stride).
// ISA layout: lane%16 = row, kb=(lane/16)*8; VGPR0..3 K=kb..kb+7, VGPR4..7 K=kb+16..kb+23.
__device__ __forceinline__ v16bf load_frag_f32(const float* __restrict__ base,
                                               int ld, int lane) {
    int row = lane & 15;
    int kb  = (lane >> 4) << 3;
    const float* p = base + row * ld + kb;
    v8u t;
#pragma unroll
    for (int i = 0; i < 4; ++i) t[i]     = pkbf(p[2*i],      p[2*i + 1]);
#pragma unroll
    for (int i = 0; i < 4; ++i) t[4 + i] = pkbf(p[16 + 2*i], p[17 + 2*i]);
    return __builtin_bit_cast(v16bf, t);
}

// Same fragment from row-major *bf16* source: two raw 16-byte loads, no VALU.
__device__ __forceinline__ v16bf load_frag_bf(const __bf16* __restrict__ base,
                                              int ld, int lane) {
    int row = lane & 15;
    int kb  = (lane >> 4) << 3;
    const __bf16* p = base + row * ld + kb;
    union { v4u u[2]; v16bf v; } cv;
    cv.u[0] = *(const v4u*)(p);
    cv.u[1] = *(const v4u*)(p + 16);
    return cv.v;
}

// f32 -> bf16 array conversion (packed, 2 elems/thread). n must be even.
__global__ void cvt_bf16(const float* __restrict__ in, __bf16* __restrict__ out, int n) {
    int i = (blockIdx.x * blockDim.x + threadIdx.x) * 2;
    if (i >= n) return;
    *(unsigned*)(out + i) = pkbf(in[i], in[i + 1]);
}

// ---------------------------------------------------------------------------
// Projection GEMM: C[M x NT*16] = A[M x K] @ W[NT*16 x K]^T + bias.
// One wave per 16-row tile holds all NT n-tile accumulators in registers; A
// fragment converted once per K-chunk; B fragments issued in groups of 4 so
// their loads overlap the preceding WMMAs.
// ---------------------------------------------------------------------------
template <int NT>
__global__ __launch_bounds__(32) void gemm_proj(
    const float* __restrict__ A, int lda,
    const __bf16* __restrict__ W, const float* __restrict__ bias,
    float* __restrict__ C, int ldc, int K) {
    int mt   = blockIdx.x;
    int lane = threadIdx.x;
    const float* Arow = A + (size_t)mt * 16 * lda;

    v8f acc[NT];
#pragma unroll
    for (int j = 0; j < NT; ++j) acc[j] = (v8f){0.f,0.f,0.f,0.f,0.f,0.f,0.f,0.f};

    for (int k0 = 0; k0 < K; k0 += 32) {
        if (k0 + 32 < K) __builtin_prefetch(Arow + k0 + 32, 0, 1);
        v16bf a = load_frag_f32(Arow + k0, lda, lane);
#pragma unroll
        for (int jg = 0; jg < NT; jg += 4) {
            v16bf b[4];
#pragma unroll
            for (int q = 0; q < 4; ++q)
                if (jg + q < NT)
                    b[q] = load_frag_bf(W + (size_t)(jg + q) * 16 * K + k0, K, lane);
#pragma unroll
            for (int q = 0; q < 4; ++q)
                if (jg + q < NT)
                    acc[jg + q] = __builtin_amdgcn_wmma_f32_16x16x32_bf16(
                        false, a, false, b[q], (short)0, acc[jg + q], false, false);
        }
    }
    int nl = lane & 15;
    int mb = (lane >> 4) << 3;
#pragma unroll
    for (int j = 0; j < NT; ++j) {
        float bv = bias[j * 16 + nl];
#pragma unroll
        for (int v = 0; v < 8; ++v) {
            size_t m = (size_t)mt * 16 + mb + v;
            C[m * ldc + j * 16 + nl] = acc[j][v] + bv;
        }
    }
}

// ---------------------------------------------------------------------------
// Rate GEMM (K == 32, huge N): A fragment converted once, then stream n-tiles
// in groups of 4: bf16 B loads -> wmma -> biased NON-TEMPORAL stores (256 MB
// of logits is write-once; keep it out of L2). grid = (M/16, N/1024).
// ---------------------------------------------------------------------------
__global__ __launch_bounds__(32) void gemm_k32(
    const float* __restrict__ A,            // (M,32) f32
    const __bf16* __restrict__ W,           // (N,32) bf16
    const float* __restrict__ bias,
    float* __restrict__ C, int ldc) {
    int mt   = blockIdx.x;
    int lane = threadIdx.x;
    int nl   = lane & 15;
    int mb   = (lane >> 4) << 3;

    v16bf a = load_frag_f32(A + (size_t)mt * 16 * 32, 32, lane);

    int nbase = blockIdx.y * 64;            // 64 n-tiles of 16 per block
    for (int nt = 0; nt < 64; nt += 4) {
        v16bf b[4];
#pragma unroll
        for (int q = 0; q < 4; ++q)
            b[q] = load_frag_bf(W + (size_t)(nbase + nt + q) * 16 * 32, 32, lane);
#pragma unroll
        for (int q = 0; q < 4; ++q) {
            int n0 = (nbase + nt + q) * 16;
            v8f acc = (v8f){0.f,0.f,0.f,0.f,0.f,0.f,0.f,0.f};
            acc = __builtin_amdgcn_wmma_f32_16x16x32_bf16(
                false, a, false, b[q], (short)0, acc, false, false);
            float bv = bias[n0 + nl];
#pragma unroll
            for (int v = 0; v < 8; ++v) {
                size_t m = (size_t)mt * 16 + mb + v;
                __builtin_nontemporal_store(acc[v] + bv, &C[m * ldc + n0 + nl]);
            }
        }
    }
}

// ---------------------------------------------------------------------------
// Persistent GRU recurrence: one block (8 waves) per direction-instance.
// h kept f32 in LDS; Whh held as bf16 B-fragments in VGPRs for all T steps.
// ---------------------------------------------------------------------------
struct GruJob {
    const float* xp;    // (B*T,192) input projections (incl. bih) or nullptr
    const float* Whh;   // (192,64)
    const float* bih;   // (192)  used only when xp == nullptr (u == 0 pass)
    const float* bhh;   // (192)
    const float* h0;    // (32,64) or nullptr (zeros)
    float* y;           // (B*T, ystride) or nullptr
    float* hT;          // (32,64) or nullptr
    int ystride, yoff, rev;
};

__global__ __launch_bounds__(256) void gru_seq(GruJob j0, GruJob j1,
                                               GruJob j2, GruJob j3) {
    GruJob J;
    switch (blockIdx.x) {
        case 0:  J = j0; break;
        case 1:  J = j1; break;
        case 2:  J = j2; break;
        default: J = j3; break;
    }
    __shared__ float hls[BN * HS];
    __shared__ float ghls[BN * GS];
    int tid  = threadIdx.x;
    int lane = tid & 31;
    int wave = tid >> 5;
    int mtile = wave >> 2;            // waves 0-3 -> rows 0-15, 4-7 -> 16-31
    int ng    = (wave & 3) * 3;       // first of this wave's 3 n-tiles

    for (int i = tid; i < BN * HH; i += 256)
        hls[(i >> 6) * HS + (i & 63)] = J.h0 ? J.h0[i] : 0.0f;

    v16bf Bf[3][2];                   // persistent Whh fragments (bf16)
#pragma unroll
    for (int j = 0; j < 3; ++j)
#pragma unroll
        for (int c = 0; c < 2; ++c)
            Bf[j][c] = load_frag_f32(J.Whh + (size_t)(ng + j) * 16 * HH + c * 32, HH, lane);
    __syncthreads();

    for (int s = 0; s < TT; ++s) {
        int t = J.rev ? (TT - 1 - s) : s;
        v16bf a0 = load_frag_f32(hls + mtile * 16 * HS + 0,  HS, lane);
        v16bf a1 = load_frag_f32(hls + mtile * 16 * HS + 32, HS, lane);
        int nl = lane & 15;
        int mb = (lane >> 4) << 3;
#pragma unroll
        for (int j = 0; j < 3; ++j) {
            v8f acc = (v8f){0.f,0.f,0.f,0.f,0.f,0.f,0.f,0.f};
            acc = __builtin_amdgcn_wmma_f32_16x16x32_bf16(false, a0, false, Bf[j][0],
                                                          (short)0, acc, false, false);
            acc = __builtin_amdgcn_wmma_f32_16x16x32_bf16(false, a1, false, Bf[j][1],
                                                          (short)0, acc, false, false);
            int n0 = (ng + j) * 16;
#pragma unroll
            for (int v = 0; v < 8; ++v)
                ghls[(mtile * 16 + mb + v) * GS + n0 + nl] = acc[v];
        }
        __syncthreads();
        for (int i = tid; i < BN * HH; i += 256) {
            int bb = i >> 6, u = i & 63;
            float hr = ghls[bb * GS + u]        + J.bhh[u];
            float hz = ghls[bb * GS + 64 + u]   + J.bhh[64 + u];
            float hn = ghls[bb * GS + 128 + u]  + J.bhh[128 + u];
            float ir, iz, inn;
            if (J.xp) {
                const float* xr = J.xp + ((size_t)bb * TT + t) * G3;
                ir = xr[u]; iz = xr[64 + u]; inn = xr[128 + u];
            } else {
                ir = J.bih[u]; iz = J.bih[64 + u]; inn = J.bih[128 + u];
            }
            float r  = 1.0f / (1.0f + __expf(-(ir + hr)));
            float z  = 1.0f / (1.0f + __expf(-(iz + hz)));
            float nn = tanhf(inn + r * hn);
            float ho = hls[bb * HS + u];
            float hv = (1.0f - z) * nn + z * ho;
            hls[bb * HS + u] = hv;
            if (J.y) J.y[((size_t)bb * TT + t) * J.ystride + J.yoff + u] = hv;
        }
        __syncthreads();
    }
    if (J.hT)
        for (int i = tid; i < BN * HH; i += 256)
            J.hT[i] = hls[(i >> 6) * HS + (i & 63)];
}

// ---------------------------------------------------------------------------
// Small fused kernels
// ---------------------------------------------------------------------------
__global__ __launch_bounds__(256) void g0_sample(
    const float* __restrict__ hF, const float* __restrict__ hB,
    const float* __restrict__ W,  const float* __restrict__ b,
    const float* __restrict__ eps,
    const float* __restrict__ lfW, const float* __restrict__ lfb,
    const float* __restrict__ lbW, const float* __restrict__ lbb,
    float* __restrict__ mu_out, float* __restrict__ lv_out,
    float* __restrict__ h0f, float* __restrict__ h0b) {
    __shared__ float g0p[32 * 32];
    __shared__ float g0s[32 * 16];
    int tid = threadIdx.x;
    for (int i = tid; i < 32 * 32; i += 256) {          // g0p = [hF,hB] @ W^T + b
        int bb = i >> 5, o = i & 31;
        float s = b[o];
        for (int k = 0; k < 64; ++k) s += hF[bb * 64 + k] * W[o * 128 + k];
        for (int k = 0; k < 64; ++k) s += hB[bb * 64 + k] * W[o * 128 + 64 + k];
        g0p[i] = s;
    }
    __syncthreads();
    for (int i = tid; i < 32 * 16; i += 256) {          // reparameterized sample
        int bb = i >> 4, l = i & 15;
        float mu = g0p[bb * 32 + l], lv = g0p[bb * 32 + 16 + l];
        mu_out[i] = mu; lv_out[i] = lv;
        g0s[i] = mu + eps[i] * __expf(0.5f * lv);
    }
    __syncthreads();
    for (int i = tid; i < 32 * 64; i += 256) {          // h0f/h0b linears
        int bb = i >> 6, h = i & 63;
        float sf = lfb[h], sb = lbb[h];
        for (int k = 0; k < 8; ++k) {
            sf += g0s[bb * 16 + k]     * lfW[h * 8 + k];
            sb += g0s[bb * 16 + 8 + k] * lbW[h * 8 + k];
        }
        h0f[i] = sf; h0b[i] = sb;
    }
}

__global__ void u_sample(const float* __restrict__ up, const float* __restrict__ eps,
                         float* __restrict__ mu, float* __restrict__ lv,
                         float* __restrict__ u, int n) {
    int i = blockIdx.x * blockDim.x + threadIdx.x;
    if (i >= n) return;
    int row = i >> 3, c = i & 7;
    float m = up[row * 16 + c], l = up[row * 16 + 8 + c];
    mu[i] = m; lv[i] = l;
    u[i] = m + eps[i] * __expf(0.5f * l);
}

__global__ void ctrl_pack(const float* __restrict__ enc, const float* __restrict__ f_init,
                          float* __restrict__ out, int total) {
    int i = blockIdx.x * blockDim.x + threadIdx.x;
    if (i >= total) return;
    int row = i / 160, c = i % 160;
    float v;
    if (c < 128) v = enc[(size_t)row * 128 + c];
    else {
        int t = row % TT;                       // row = b*T + t
        v = (t == 0) ? 0.0f : f_init[(size_t)(row - 1) * 32 + (c - 128)];
    }
    out[i] = v;
}

__global__ void gen2_proj(const float* __restrict__ u,
                          const float* __restrict__ Wf, const float* __restrict__ bf,
                          const float* __restrict__ Wb, const float* __restrict__ bb,
                          float* __restrict__ xpf, float* __restrict__ xpb, int per) {
    int i = blockIdx.x * blockDim.x + threadIdx.x;
    if (i >= 2 * per) return;
    int d = i / per, r = i - d * per;
    int row = r / G3, o = r - row * G3;
    const float* W  = d ? Wb : Wf;
    const float* bi = d ? bb : bf;
    float s = bi[o];
    const float* ur = u + (size_t)row * 8;
#pragma unroll
    for (int k = 0; k < 8; ++k) s += ur[k] * W[o * 8 + k];
    (d ? xpb : xpf)[r] = s;
}

// ---------------------------------------------------------------------------
// Host orchestration
// ---------------------------------------------------------------------------
extern "C" void kernel_launch(void* const* d_in, const int* in_sizes, int n_in,
                              void* d_out, int out_size, void* d_ws, size_t ws_size,
                              hipStream_t stream) {
    (void)in_sizes; (void)n_in; (void)out_size; (void)ws_size;
    auto in = [&](int i) { return (const float*)d_in[i]; };
    const float* x      = in(0);
    const float* eps_g0 = in(1);
    const float* eps_u  = in(2);

    float* ws = (float*)d_ws;
    const size_t XP = (size_t)BT * G3;
    float* xp0 = ws;               // g0f xp -> later ctrl_in
    float* xp1 = ws + XP;          // g0b xp -> later ctrl xp
    float* xp2 = ws + 2 * XP;      // cef xp -> later genf xp (pass2)
    float* xp3 = ws + 3 * XP;      // ceb xp -> later genb xp (pass2)
    float* enc_out  = ws + 4 * XP;                       // BT*128
    float* gen_out  = enc_out + (size_t)BT * 128;        // BT*128 (both passes)
    float* f_init   = gen_out + (size_t)BT * 128;        // BT*32
    float* ctrl_out = f_init  + (size_t)BT * 32;         // BT*64
    float* up       = ctrl_out + (size_t)BT * 64;        // BT*16
    float* uu       = up + (size_t)BT * 16;              // BT*8
    float* hF  = uu  + (size_t)BT * 8;
    float* hB  = hF  + 2048;
    float* h0f = hB  + 2048;
    float* h0b = h0f + 2048;

    // bf16 weight staging area (starts 16B-aligned: all offsets are even)
    __bf16* wb      = (__bf16*)(h0b + 2048);
    const int WIHN  = G3 * 4096;                         // 786432
    __bf16* wih_bf[4] = { wb, wb + WIHN, wb + 2*WIHN, wb + 3*WIHN };
    __bf16* ctrl_bf = wb + 4*WIHN;                       // 192*160
    __bf16* fac_bf  = ctrl_bf + G3*160;                  // 32*128
    __bf16* ulin_bf = fac_bf + 32*128;                   // 16*64
    __bf16* rate_bf = ulin_bf + 16*64;                   // 4096*32

    float* out    = (float*)d_out;
    float* logits = out;                                  // (B,T,4096)
    float* f_t    = out + (size_t)BT * 4096;              // (B,T,32)
    float* mu_g0  = f_t + (size_t)BT * 32;                // (B,16)
    float* lv_g0  = mu_g0 + 512;                          // (B,16)
    float* mu_u   = lv_g0 + 512;                          // (B,T,8)
    float* lv_u   = mu_u + (size_t)BT * 8;                // (B,T,8)

    dim3 w32(32);
    auto cvt = [&](const float* src, __bf16* dst, int n) {
        cvt_bf16<<<(n/2 + 255) / 256, 256, 0, stream>>>(src, dst, n);
    };

    // Stage 0: convert weights to bf16 once
    cvt(in(3),  wih_bf[0], WIHN);
    cvt(in(7),  wih_bf[1], WIHN);
    cvt(in(11), wih_bf[2], WIHN);
    cvt(in(15), wih_bf[3], WIHN);
    cvt(in(19), ctrl_bf, G3*160);
    cvt(in(39), fac_bf,  32*128);
    cvt(in(33), ulin_bf, 16*64);
    cvt(in(41), rate_bf, 4096*32);

    // Phase A: 4 encoder input projections (16384x4096 @ 4096x192)
    gemm_proj<12><<<BT/16, w32, 0, stream>>>(x, 4096, wih_bf[0], in(5),  xp0, G3, 4096);
    gemm_proj<12><<<BT/16, w32, 0, stream>>>(x, 4096, wih_bf[1], in(9),  xp1, G3, 4096);
    gemm_proj<12><<<BT/16, w32, 0, stream>>>(x, 4096, wih_bf[2], in(13), xp2, G3, 4096);
    gemm_proj<12><<<BT/16, w32, 0, stream>>>(x, 4096, wih_bf[3], in(17), xp3, G3, 4096);

    // Phase B: 4 encoder recurrences concurrently (block per direction)
    GruJob e0{ xp0, in(4),  in(5),  in(6),  nullptr, nullptr, hF, 128, 0,  0 };
    GruJob e1{ xp1, in(8),  in(9),  in(10), nullptr, nullptr, hB, 128, 0,  1 };
    GruJob e2{ xp2, in(12), in(13), in(14), nullptr, enc_out, nullptr, 128, 0,  0 };
    GruJob e3{ xp3, in(16), in(17), in(18), nullptr, enc_out, nullptr, 128, 64, 1 };
    gru_seq<<<4, 256, 0, stream>>>(e0, e1, e2, e3);

    // g0 sample + generator initial states
    g0_sample<<<1, 256, 0, stream>>>(hF, hB, in(31), in(32), eps_g0,
                                     in(35), in(36), in(37), in(38),
                                     mu_g0, lv_g0, h0f, h0b);

    // Phase C: generator pass 1 (u == 0 -> xp = bih)
    GruJob g1f{ nullptr, in(24), in(25), in(26), h0f, gen_out, nullptr, 128, 0,  0 };
    GruJob g1b{ nullptr, in(28), in(29), in(30), h0b, gen_out, nullptr, 128, 64, 1 };
    gru_seq<<<2, 256, 0, stream>>>(g1f, g1b, g1f, g1f);
    gemm_proj<2><<<BT/16, w32, 0, stream>>>(gen_out, 128, fac_bf, in(40), f_init, 32, 128);

    // Phase D: controller
    ctrl_pack<<<(BT*160 + 255)/256, 256, 0, stream>>>(enc_out, f_init, xp0, BT*160);
    gemm_proj<12><<<BT/16, w32, 0, stream>>>(xp0, 160, ctrl_bf, in(21), xp1, G3, 160);
    GruJob c0{ xp1, in(20), in(21), in(22), nullptr, ctrl_out, nullptr, 64, 0, 0 };
    gru_seq<<<1, 256, 0, stream>>>(c0, c0, c0, c0);
    gemm_proj<1><<<BT/16, w32, 0, stream>>>(ctrl_out, 64, ulin_bf, in(34), up, 16, 64);
    u_sample<<<(BT*8 + 255)/256, 256, 0, stream>>>(up, eps_u, mu_u, lv_u, uu, BT*8);

    // Phase E: generator pass 2 + factors + rates
    gen2_proj<<<(2*BT*G3 + 255)/256, 256, 0, stream>>>(uu, in(23), in(25), in(27), in(29),
                                                       xp2, xp3, BT*G3);
    GruJob g2f{ xp2, in(24), in(25), in(26), h0f, gen_out, nullptr, 128, 0,  0 };
    GruJob g2b{ xp3, in(28), in(29), in(30), h0b, gen_out, nullptr, 128, 64, 1 };
    gru_seq<<<2, 256, 0, stream>>>(g2f, g2b, g2f, g2f);
    gemm_proj<2><<<BT/16, w32, 0, stream>>>(gen_out, 128, fac_bf, in(40), f_t, 32, 128);
    gemm_k32<<<dim3(BT/16, 4), w32, 0, stream>>>(f_t, rate_bf, in(42), logits, 4096);
}
